// GCNReg_binary_add_33243046871481
// MI455X (gfx1250) — compile-verified, matrix operand order checked by reading, not run on hardware
//
#include <hip/hip_runtime.h>
#include <hip/hip_bf16.h>
#include <stdint.h>

// ---------------------------------------------------------------------------
// GCNReg for MI455X (gfx1250, wave32).  WMMA bf16 GEMMs + L2-resident
// edge scatter with f32 atomics.
// ---------------------------------------------------------------------------

typedef __attribute__((ext_vector_type(16))) __bf16 v16bf;
typedef __attribute__((ext_vector_type(8)))  float  v8f;

#define N_NODES 50000
#define N_EDGES 500000
#define NB      512
#define DIM_IN  128
#define DIM_H   128
#define DIM_EX  64
#define DIM_CAT (2 * DIM_H + DIM_EX)   // 320

// f32 -> bf16 round-to-nearest-even, packed pair -> u32
__device__ __forceinline__ unsigned int f2bf(float f) {
  unsigned int u = __float_as_uint(f);
  return (u + 0x7FFFu + ((u >> 16) & 1u)) >> 16;
}
__device__ __forceinline__ unsigned int pk2bf(float lo, float hi) {
  return f2bf(lo) | (f2bf(hi) << 16);
}

// ---------------------------------------------------------------------------
// Tiled WMMA GEMM:  C[M,Nw] = op( A[M,K] (row-scaled) @ B[K,Nw] + bias )
//   A, B f32 row-major; bf16 staged in double-buffered LDS; f32 accumulate.
//   rowscale: optional per-row multiplier on A (deg^-1/2); bias/relu epilogue.
//   Requires K % 32 == 0 and Nw % 16 == 0 (true at all call sites).
// Block: 256 threads = 8 waves.  Block tile 128x128, wave tile 16x128.
// Partial N-tiles are zero-padded in LDS -> WMMA loop is branch-free,
// EXEC stays all-ones through every v_wmma (ISA requirement).
// ---------------------------------------------------------------------------
#define TM   128
#define TN   128
#define KT   32
#define LDSP 40   // padded row stride in ushorts (80B = 20 banks, conflict-free)

__global__ void __launch_bounds__(256)
wmma_gemm(const float* __restrict__ A, const float* __restrict__ Bm,
          float* __restrict__ C, const float* __restrict__ rowscale,
          const float* __restrict__ bias, int M, int K, int Nw, int relu)
{
  __shared__ unsigned short ldsA[2][TM * LDSP];  // [buf][row][k]  128 x 32 bf16
  __shared__ unsigned short ldsB[2][TN * LDSP];  // [buf][col][k]  transposed

  const int tid  = threadIdx.x;
  const int wave = tid >> 5;
  const int lane = tid & 31;
  const int l16  = lane & 15;
  const int hi   = lane >> 4;          // 0: K {0..7,16..23}, 1: K {8..15,24..31}

  const int rowBase = blockIdx.x * TM;
  const int colBase = blockIdx.y * TN;

  // ---- staging thread mappings (fixed per thread) ----
  const int ar  = tid >> 1;            // A: row 0..127
  const int ac0 = (tid & 1) << 4;      // A: k offset 0 or 16 (16 floats)
  const int bc0 = (tid & 15) << 3;     // B: col offset, 8 cols
  const int bk  = (tid >> 4) << 1;     // B: k pair base 0,2,..,30
  const bool bOK = (colBase + bc0 + 8) <= Nw;  // whole 8-col group in-bounds
                                               // (Nw % 8 == 0 at all call sites)

  v8f acc[8] = {};

  // ---- tile stagers ----
  auto stageA = [&](int buf, int kk) {
    int grow = rowBase + ar;
    float v[16];
    if (grow < M) {
      float s = rowscale ? rowscale[grow] : 1.0f;
      const float4* sp = (const float4*)(A + (size_t)grow * K + kk + ac0);
      __builtin_prefetch((const float*)sp + 2 * KT, 0, 1);  // ~2 K-steps ahead
      float4 q0 = sp[0], q1 = sp[1], q2 = sp[2], q3 = sp[3];
      v[0]=q0.x*s;  v[1]=q0.y*s;  v[2]=q0.z*s;  v[3]=q0.w*s;
      v[4]=q1.x*s;  v[5]=q1.y*s;  v[6]=q1.z*s;  v[7]=q1.w*s;
      v[8]=q2.x*s;  v[9]=q2.y*s;  v[10]=q2.z*s; v[11]=q2.w*s;
      v[12]=q3.x*s; v[13]=q3.y*s; v[14]=q3.z*s; v[15]=q3.w*s;
    } else {
#pragma unroll
      for (int j = 0; j < 16; ++j) v[j] = 0.0f;
    }
    unsigned int p[8];
#pragma unroll
    for (int j = 0; j < 8; ++j) p[j] = pk2bf(v[2 * j], v[2 * j + 1]);
    uint4* dp = (uint4*)&ldsA[buf][ar * LDSP + ac0];
    dp[0] = make_uint4(p[0], p[1], p[2], p[3]);
    dp[1] = make_uint4(p[4], p[5], p[6], p[7]);
  };
  auto stageB = [&](int buf, int kk) {
    unsigned int p[8];
    if (bOK) {                               // uniform per thread, one branch
      const float4* r0 = (const float4*)(Bm + (size_t)(kk + bk) * Nw + colBase + bc0);
      const float4* r1 = (const float4*)((const float*)r0 + Nw);
      float4 a0 = r0[0], a1 = r0[1];
      float4 b0 = r1[0], b1 = r1[1];
      p[0] = pk2bf(a0.x, b0.x); p[1] = pk2bf(a0.y, b0.y);
      p[2] = pk2bf(a0.z, b0.z); p[3] = pk2bf(a0.w, b0.w);
      p[4] = pk2bf(a1.x, b1.x); p[5] = pk2bf(a1.y, b1.y);
      p[6] = pk2bf(a1.z, b1.z); p[7] = pk2bf(a1.w, b1.w);
    } else {
#pragma unroll
      for (int j = 0; j < 8; ++j) p[j] = 0u;
    }
#pragma unroll
    for (int j = 0; j < 8; ++j)              // packed (k,k+1) pair per column
      *(unsigned int*)&ldsB[buf][(bc0 + j) * LDSP + bk] = p[j];
  };

  stageA(0, 0);
  stageB(0, 0);
  __syncthreads();

  int s = 0;
  for (int kk = 0; kk < K; kk += KT, s ^= 1) {
    if (kk + KT < K) {                 // stage next tile into other buffer
      stageA(s ^ 1, kk + KT);
      stageB(s ^ 1, kk + KT);
    }
    // ---- fragments per CDNA5 16-bit A/B VGPR layout; branch-free WMMA ----
    union FU { uint4 q[2]; v16bf v; } fa, fb;
    const uint4* pa = (const uint4*)&ldsA[s][(wave * 16 + l16) * LDSP];
    fa.q[0] = pa[hi];                  // K  hi*8 .. hi*8+7
    fa.q[1] = pa[2 + hi];              // K 16+hi*8 .. 16+hi*8+7
#pragma unroll
    for (int n = 0; n < 8; ++n) {
      const uint4* pb = (const uint4*)&ldsB[s][(n * 16 + l16) * LDSP];
      fb.q[0] = pb[hi];
      fb.q[1] = pb[2 + hi];
      acc[n] = __builtin_amdgcn_wmma_f32_16x16x32_bf16(
          false, fa.v, false, fb.v, (short)0, acc[n], false, false);
    }
    __syncthreads();
  }

  // ---- epilogue: C/D layout -> row = base + i + hi*8, col = base + l16 ----
  for (int n = 0; n < 8; ++n) {
    int col = colBase + n * 16 + l16;  // col guard uniform per n (Nw % 16 == 0)
    if (col >= Nw) continue;
    float bv = bias ? bias[col] : 0.0f;
#pragma unroll
    for (int i = 0; i < 8; ++i) {
      int row = rowBase + wave * 16 + hi * 8 + i;
      if (row >= M) continue;
      float v = acc[n][i] + bv;
      if (relu) v = fmaxf(v, 0.0f);
      C[(size_t)row * Nw + col] = v;
    }
  }
}

// ---------------------------------------------------------------------------
// Small helper kernels
// ---------------------------------------------------------------------------
__global__ void zero_u32(unsigned int* p, size_t n) {
  size_t i = (size_t)blockIdx.x * blockDim.x + threadIdx.x;
  if (i < n) p[i] = 0u;
}

__global__ void degree_count(const int* __restrict__ src, const int* __restrict__ dst,
                             int* dout, int* din, int E) {
  int i = blockIdx.x * blockDim.x + threadIdx.x;
  if (i < E) {
    atomicAdd(&dout[src[i]], 1);
    atomicAdd(&din[dst[i]], 1);
  }
}

__global__ void make_rs(const int* __restrict__ dout, const int* __restrict__ din,
                        float* rs_out, float* rs_in, int n) {
  int i = blockIdx.x * blockDim.x + threadIdx.x;
  if (i < n) {
    rs_out[i] = rsqrtf(fmaxf((float)dout[i], 1.0f));
    rs_in[i]  = rsqrtf(fmaxf((float)din[i], 1.0f));
  }
}

// one wave32 per edge: lane covers 4 floats via float4 + 4 atomic f32 adds
__global__ void scatter_add(const float* __restrict__ t, const int* __restrict__ src,
                            const int* __restrict__ dst, float* __restrict__ agg, int E) {
  int e    = blockIdx.x * (blockDim.x >> 5) + (threadIdx.x >> 5);
  int lane = threadIdx.x & 31;
  if (e >= E) return;
  int sidx = src[e], didx = dst[e];
  float4 v = ((const float4*)(t + (size_t)sidx * DIM_H))[lane];
  float* dp = agg + (size_t)didx * DIM_H + lane * 4;
  atomicAdd(dp + 0, v.x);
  atomicAdd(dp + 1, v.y);
  atomicAdd(dp + 2, v.z);
  atomicAdd(dp + 3, v.w);
}

// h = relu(h * rs_in[row] + bias[col]), in place over N*H elements
__global__ void finalize_layer(float* __restrict__ h, const float* __restrict__ rs_in,
                               const float* __restrict__ bias, int n) {
  int i = blockIdx.x * blockDim.x + threadIdx.x;
  if (i < n) {
    int row = i >> 7, col = i & 127;
    h[i] = fmaxf(h[i] * rs_in[row] + bias[col], 0.0f);
  }
}

// one wave32 per node: segment sums + counts
__global__ void pool_accum(const float* __restrict__ h, const int* __restrict__ seg,
                           float* __restrict__ sums, int* __restrict__ cnt, int n) {
  int i    = blockIdx.x * (blockDim.x >> 5) + (threadIdx.x >> 5);
  int lane = threadIdx.x & 31;
  if (i >= n) return;
  int s = seg[i];
  float4 v = ((const float4*)(h + (size_t)i * DIM_H))[lane];
  float* dp = sums + (size_t)s * DIM_H + lane * 4;
  atomicAdd(dp + 0, v.x);
  atomicAdd(dp + 1, v.y);
  atomicAdd(dp + 2, v.z);
  atomicAdd(dp + 3, v.w);
  if (lane == 0) atomicAdd(&cnt[s], 1);
}

__global__ void pool_write(const float* __restrict__ sums, const int* __restrict__ cnt,
                           float* __restrict__ hg, int colOff) {
  int i = blockIdx.x * blockDim.x + threadIdx.x;
  if (i < NB * DIM_H) {
    int b = i >> 7, c = i & 127;
    hg[(size_t)b * DIM_CAT + colOff + c] = sums[i] / fmaxf((float)cnt[b], 1.0f);
  }
}

__global__ void desc_copy(const float* __restrict__ desc, float* __restrict__ hg) {
  int i = blockIdx.x * blockDim.x + threadIdx.x;
  if (i < NB * DIM_EX) {
    int b = i / DIM_EX, c = i - b * DIM_EX;
    hg[(size_t)b * DIM_CAT + 2 * DIM_H + c] = desc[i];
  }
}

// out[b] = o3[b,:] . Wc4[:,0] + bc4
__global__ void head_out(const float* __restrict__ o3, const float* __restrict__ Wc4,
                         const float* __restrict__ bc4, float* __restrict__ out) {
  int b = blockIdx.x * blockDim.x + threadIdx.x;
  if (b < NB) {
    const float4* r = (const float4*)(o3 + (size_t)b * DIM_H);
    const float4* w = (const float4*)Wc4;
    float acc = 0.0f;
#pragma unroll
    for (int j = 0; j < DIM_H / 4; ++j) {
      float4 a = r[j], c = w[j];
      acc += a.x * c.x + a.y * c.y + a.z * c.z + a.w * c.w;
    }
    out[b] = acc + bc4[0];
  }
}

// ---------------------------------------------------------------------------
// Host orchestration (graph-capture safe: kernels only, all on `stream`)
// ---------------------------------------------------------------------------
static inline unsigned div_up(size_t n, int t) { return (unsigned)((n + t - 1) / t); }

extern "C" void kernel_launch(void* const* d_in, const int* in_sizes, int n_in,
                              void* d_out, int out_size, void* d_ws, size_t ws_size,
                              hipStream_t stream)
{
  (void)in_sizes; (void)n_in; (void)out_size; (void)ws_size;

  const float* h1   = (const float*)d_in[0];
  const float* h2   = (const float*)d_in[1];
  const float* desc = (const float*)d_in[2];
  const int*   e1   = (const int*)d_in[3];
  const int*   e2   = (const int*)d_in[4];
  const int*   sg1  = (const int*)d_in[5];
  const int*   sg2  = (const int*)d_in[6];
  const float* W1  = (const float*)d_in[7];
  const float* b1  = (const float*)d_in[8];
  const float* W2  = (const float*)d_in[9];
  const float* b2  = (const float*)d_in[10];
  const float* Wc1 = (const float*)d_in[11];
  const float* bc1 = (const float*)d_in[12];
  const float* Wc2 = (const float*)d_in[13];
  const float* bc2 = (const float*)d_in[14];
  const float* Wc3 = (const float*)d_in[15];
  const float* bc3 = (const float*)d_in[16];
  const float* Wc4 = (const float*)d_in[17];
  const float* bc4 = (const float*)d_in[18];
  float* out = (float*)d_out;

  // ---- workspace carve (256B aligned) ----
  uintptr_t base = (uintptr_t)d_ws;
  size_t off = 0;
  auto carve = [&](size_t bytes) -> void* {
    void* p = (void*)(base + off);
    off += (bytes + 255) & ~(size_t)255;
    return p;
  };
  float* bufA   = (float*)carve((size_t)N_NODES * DIM_H * sizeof(float)); // agg / h
  float* bufB   = (float*)carve((size_t)N_NODES * DIM_H * sizeof(float)); // t = hW
  int*   degb   = (int*)  carve((size_t)2 * N_NODES * sizeof(int));
  float* rs_out = (float*)carve((size_t)N_NODES * sizeof(float));
  float* rs_in  = (float*)carve((size_t)N_NODES * sizeof(float));
  float* sums   = (float*)carve((size_t)NB * DIM_H * sizeof(float) + NB * sizeof(int));
  int*   cnt    = (int*)(sums + (size_t)NB * DIM_H);
  float* hg     = (float*)carve((size_t)NB * DIM_CAT * sizeof(float));
  float* o1     = (float*)carve((size_t)NB * DIM_CAT * sizeof(float));
  float* o2     = (float*)carve((size_t)NB * DIM_H * sizeof(float));
  float* o3     = (float*)carve((size_t)NB * DIM_H * sizeof(float));
  int* dout = degb;
  int* din  = degb + N_NODES;

  const int T = 256;
  const float* hin[2] = { h1, h2 };
  const int*   ed[2]  = { e1, e2 };
  const int*   sg[2]  = { sg1, sg2 };

  const dim3 gNode((N_NODES + TM - 1) / TM, 1);                 // 391 x 1
  const size_t NH = (size_t)N_NODES * DIM_H;

  for (int g = 0; g < 2; ++g) {
    const int* src = ed[g];
    const int* dst = ed[g] + N_EDGES;

    // degrees -> rsqrt scales
    zero_u32<<<div_up(2 * N_NODES, T), T, 0, stream>>>((unsigned*)degb, 2 * N_NODES);
    degree_count<<<div_up(N_EDGES, T), T, 0, stream>>>(src, dst, dout, din, N_EDGES);
    make_rs<<<div_up(N_NODES, T), T, 0, stream>>>(dout, din, rs_out, rs_in, N_NODES);

    // layer 1: t = (h .* rs_out) @ W1 ; agg = scatter ; h = relu(agg .* rs_in + b1)
    wmma_gemm<<<gNode, T, 0, stream>>>(hin[g], W1, bufB, rs_out, nullptr,
                                       N_NODES, DIM_IN, DIM_H, 0);
    zero_u32<<<div_up(NH, T), T, 0, stream>>>((unsigned*)bufA, NH);
    scatter_add<<<div_up(N_EDGES, 8), T, 0, stream>>>(bufB, src, dst, bufA, N_EDGES);
    finalize_layer<<<div_up(NH, T), T, 0, stream>>>(bufA, rs_in, b1, (int)NH);

    // layer 2
    wmma_gemm<<<gNode, T, 0, stream>>>(bufA, W2, bufB, rs_out, nullptr,
                                       N_NODES, DIM_H, DIM_H, 0);
    zero_u32<<<div_up(NH, T), T, 0, stream>>>((unsigned*)bufA, NH);
    scatter_add<<<div_up(N_EDGES, 8), T, 0, stream>>>(bufB, src, dst, bufA, N_EDGES);
    finalize_layer<<<div_up(NH, T), T, 0, stream>>>(bufA, rs_in, b2, (int)NH);

    // mean pooling into hg[:, g*128 : g*128+128]
    zero_u32<<<div_up((size_t)NB * DIM_H + NB, T), T, 0, stream>>>(
        (unsigned*)sums, (size_t)NB * DIM_H + NB);
    pool_accum<<<div_up(N_NODES, 8), T, 0, stream>>>(bufA, sg[g], sums, cnt, N_NODES);
    pool_write<<<div_up((size_t)NB * DIM_H, T), T, 0, stream>>>(sums, cnt, hg, g * DIM_H);
  }

  // concat descriptors
  desc_copy<<<div_up((size_t)NB * DIM_EX, T), T, 0, stream>>>(desc, hg);

  // MLP head (WMMA GEMMs with fused bias+relu)
  wmma_gemm<<<dim3((NB + TM - 1) / TM, (DIM_CAT + TN - 1) / TN), T, 0, stream>>>(
      hg, Wc1, o1, nullptr, bc1, NB, DIM_CAT, DIM_CAT, 1);
  wmma_gemm<<<dim3((NB + TM - 1) / TM, 1), T, 0, stream>>>(
      o1, Wc2, o2, nullptr, bc2, NB, DIM_CAT, DIM_H, 1);
  wmma_gemm<<<dim3((NB + TM - 1) / TM, 1), T, 0, stream>>>(
      o2, Wc3, o3, nullptr, bc3, NB, DIM_H, DIM_H, 1);
  head_out<<<div_up(NB, T), T, 0, stream>>>(o3, Wc4, bc4, out);
}
